// CircularODE_39642548142662
// MI455X (gfx1250) — matrix operand order repaired and checked by reading, not compiled, not run on hardware
//
#include <hip/hip_runtime.h>
#include <math.h>

#define B_SZ   16384
#define T_SZ   200
#define HID    128
#define H2     64
#define WAVES  4
#define ROWS   (WAVES*16)
#define BLOCK  (WAVES*32)

typedef __attribute__((ext_vector_type(16))) _Float16 v16h;
typedef __attribute__((ext_vector_type(8)))  float    v8f;

struct alignas(32) Smem {
    _Float16 Bdr1[8][32][16];    // dr_w1 padded 32x128, B-frag layout
    _Float16 Bdr2[32][32][16];   // dr_w2 128x128, frags [kk*8+nt]
    _Float16 Bdr3[4][32][16];    // dr_w3 128x8 padded N->16
    _Float16 Bcd1[8][32][16];    // [cn_w1 | dif_w1] 8x128 padded K->32
    _Float16 Bcd2[4][32][16];    // block-diag: col0=cn_w2(rows0..63), cols1..8=dif_w2(rows64..127)
    float b1[HID], g1[HID], bt1[HID];
    float b2[HID], g2[HID], bt2[HID];
    float drb3[16];
    float cdb1[HID];             // [cn_b1 | dif_b1]
    float cnw1[8*64], cnb1[64], cnw2[64], cnb2;   // scalar path for epilogue
    float difb2[8];
    float fitw1[64], fitb1[64], fitw2[64], fitb2;
    float zbuf[ROWS][8];
    float dbuf[ROWS][16];        // shared extraction buffer (cn/dif then d3)
    alignas(16) _Float16 hbuf[WAVES][HID][16];  // col-major act tile per wave
};

struct KParams {
    const float *init, *tp, *noise;
    const float *enc_w1, *enc_b1, *enc_g, *enc_bt, *enc_w2, *enc_b2;
    const float *cn_w1, *cn_b1, *cn_w2, *cn_b2;
    const float *dr_w1, *dr_b1, *dr_g1, *dr_bt1, *dr_w2, *dr_b2, *dr_g2, *dr_bt2, *dr_w3, *dr_b3;
    const float *dif_w1, *dif_b1, *dif_w2, *dif_b2;
    const float *fit_w1, *fit_b1, *fit_w2, *fit_b2;
    const float *te_w1, *te_b1, *te_w2, *te_b2;
    float *o_traj, *o_cn, *o_final, *o_ext, *o_res, *o_var;
};

// fast transcendental helpers: avoid IEEE-exact division sequences
__device__ __forceinline__ float rcp_f(float x)     { return __builtin_amdgcn_rcpf(x); }
__device__ __forceinline__ float rsq_f(float x)     { return __builtin_amdgcn_rsqf(x); }
__device__ __forceinline__ float sqrt_f(float x)    { return __builtin_amdgcn_sqrtf(x); }
__device__ __forceinline__ float sigmoid_f(float x) { return rcp_f(1.f + __expf(-x)); }
__device__ __forceinline__ float silu_f(float x)    { return x * rcp_f(1.f + __expf(-x)); }
__device__ __forceinline__ float softplus_f(float x){ return (x > 20.f) ? x : __logf(1.f + __expf(x)); }
__device__ __forceinline__ float tanh_f(float x) {
    if (x >  10.f) return  1.f;
    if (x < -10.f) return -1.f;
    float t = __expf(2.f * x);
    return (t - 1.f) * rcp_f(t + 1.f);
}
__device__ __forceinline__ float red16(float x) {
    x += __shfl_xor(x, 1); x += __shfl_xor(x, 2);
    x += __shfl_xor(x, 4); x += __shfl_xor(x, 8);
    return x;
}
__device__ __forceinline__ v8f v8zero() {
    v8f z;
    #pragma unroll
    for (int i = 0; i < 8; ++i) z[i] = 0.f;
    return z;
}

__global__ __launch_bounds__(BLOCK) void sde_fused_kernel(KParams P) {
    extern __shared__ char smem_raw[];
    Smem& S = *reinterpret_cast<Smem*>(smem_raw);

    const int tid  = threadIdx.x;
    const int w    = tid >> 5;
    const int lane = tid & 31;
    const int col  = lane & 15;      // column within 16-wide tile (C/B layout)
    const int hi   = lane >> 4;      // lane-half
    const int rb   = hi * 8;         // C-frag row base
    const int rr   = lane & 15;      // row within wave tile (row-lane layout)
    const int zr   = w * 16 + rr;    // row within block
    const long long gRow = (long long)blockIdx.x * ROWS + zr;

    // ---------------- stage weights to LDS (f16 WMMA B-fragment layout) ----------
    for (int e = tid; e < 8*32*16; e += BLOCK) {
        int j = e & 15, ln = (e >> 4) & 31, nt = e >> 9;
        int k = ((ln >> 4) << 4) + j;
        int n = nt * 16 + (ln & 15);
        S.Bdr1[nt][ln][j] = (_Float16)((k < 25) ? P.dr_w1[k * HID + n] : 0.f);
        float cd = 0.f;
        if (k < 8) cd = (n < 64) ? P.cn_w1[k * 64 + n] : P.dif_w1[k * 64 + (n - 64)];
        S.Bcd1[nt][ln][j] = (_Float16)cd;
    }
    for (int e = tid; e < 32*32*16; e += BLOCK) {
        int j = e & 15, ln = (e >> 4) & 31, f = e >> 9;
        int kk = f >> 3, nt = f & 7;
        int k = 32 * kk + ((ln >> 4) << 4) + j;
        int n = nt * 16 + (ln & 15);
        S.Bdr2[f][ln][j] = (_Float16)P.dr_w2[k * HID + n];
    }
    for (int e = tid; e < 4*32*16; e += BLOCK) {
        int j = e & 15, ln = (e >> 4) & 31, kk = e >> 9;
        int k = 32 * kk + ((ln >> 4) << 4) + j;
        int n = ln & 15;
        S.Bdr3[kk][ln][j] = (_Float16)((n < 8) ? P.dr_w3[k * 8 + n] : 0.f);
        float v2 = 0.f;
        if (k < 64)            { if (n == 0)           v2 = P.cn_w2[k]; }
        else                   { if (n >= 1 && n <= 8) v2 = P.dif_w2[(k - 64) * 8 + (n - 1)]; }
        S.Bcd2[kk][ln][j] = (_Float16)v2;
    }
    for (int e = tid; e < HID; e += BLOCK) {
        S.b1[e] = P.dr_b1[e]; S.g1[e] = P.dr_g1[e]; S.bt1[e] = P.dr_bt1[e];
        S.b2[e] = P.dr_b2[e]; S.g2[e] = P.dr_g2[e]; S.bt2[e] = P.dr_bt2[e];
        S.cdb1[e] = (e < 64) ? P.cn_b1[e] : P.dif_b1[e - 64];
    }
    for (int e = tid; e < 8*64; e += BLOCK) S.cnw1[e] = P.cn_w1[e];
    for (int e = tid; e < 64; e += BLOCK) {
        S.cnb1[e] = P.cn_b1[e]; S.cnw2[e] = P.cn_w2[e];
        S.fitw1[e] = P.fit_w1[e]; S.fitb1[e] = P.fit_b1[e]; S.fitw2[e] = P.fit_w2[e];
    }
    if (tid < 8)  { S.drb3[tid] = P.dr_b3[tid]; S.difb2[tid] = P.dif_b2[tid]; }
    if (tid == 0) { S.cnb2 = P.cn_b2[0]; S.fitb2 = P.fit_b2[0]; }
    __syncthreads();

    // ---------------- growth (treat == 0 -> row-invariant) -----------------------
    float growth;
    {
        float acc = 0.f;
        for (int j = 0; j < H2; ++j) acc += silu_f(P.te_b1[j]) * P.te_w2[j];
        growth = tanh_f(acc + P.te_b2[0]);
    }

    // ---------------- encoder: z0 (2 lanes per row, 64 hidden each) --------------
    {
        float s0 = P.init[gRow*3+0], s1 = P.init[gRow*3+1], s2 = P.init[gRow*3+2];
        float ps = 0.f, ps2 = 0.f;
        for (int jj = 0; jj < 64; ++jj) {
            int j = hi * 64 + jj;
            float x = s0*P.enc_w1[0*HID+j] + s1*P.enc_w1[1*HID+j] + s2*P.enc_w1[2*HID+j] + P.enc_b1[j];
            ps += x; ps2 += x * x;
        }
        ps  += __shfl_xor(ps, 16);
        ps2 += __shfl_xor(ps2, 16);
        float mean = ps * (1.f/128.f);
        float rstd = rsq_f(ps2 * (1.f/128.f) - mean * mean + 1e-5f);
        float acc[8];
        #pragma unroll
        for (int o = 0; o < 8; ++o) acc[o] = 0.f;
        for (int jj = 0; jj < 64; ++jj) {
            int j = hi * 64 + jj;
            float x = s0*P.enc_w1[0*HID+j] + s1*P.enc_w1[1*HID+j] + s2*P.enc_w1[2*HID+j] + P.enc_b1[j];
            float h = silu_f((x - mean) * rstd * P.enc_g[j] + P.enc_bt[j]);
            #pragma unroll
            for (int o = 0; o < 8; ++o) acc[o] += h * P.enc_w2[j*8+o];
        }
        #pragma unroll
        for (int o = 0; o < 8; ++o) acc[o] += __shfl_xor(acc[o], 16);
        if (hi == 0) {
            #pragma unroll
            for (int o = 0; o < 8; ++o) {
                float z = acc[o] + P.enc_b2[o];
                S.zbuf[zr][o] = z;
                P.o_traj[(gRow * T_SZ + 0) * 8 + o] = z;
            }
        }
    }

    // ---------------- helpers ----------------------------------------------------
    auto cn_dec = [&](const float* zz) -> float {   // scalar path (epilogue only)
        float acc = 0.f;
        for (int jj = 0; jj < 32; ++jj) {
            int j = hi * 32 + jj;
            float a = S.cnb1[j];
            #pragma unroll
            for (int k = 0; k < 8; ++k) a += zz[k] * S.cnw1[k*64+j];
            acc += silu_f(a) * S.cnw2[j];
        }
        acc += __shfl_xor(acc, 16);
        return softplus_f(acc + S.cnb2);
    };

    auto ln_silu_store = [&](v8f* cc, const float* bb, const float* gg, const float* be) {
        float ps[8], ps2[8];
        #pragma unroll
        for (int r = 0; r < 8; ++r) { ps[r] = 0.f; ps2[r] = 0.f; }
        #pragma unroll
        for (int nt = 0; nt < 8; ++nt) {
            float bias = bb[nt*16 + col];
            #pragma unroll
            for (int r = 0; r < 8; ++r) {
                float x = cc[nt][r] + bias;
                cc[nt][r] = x; ps[r] += x; ps2[r] += x * x;
            }
        }
        float mean[8], rstd[8];
        #pragma unroll
        for (int r = 0; r < 8; ++r) {
            float s = red16(ps[r]), s2 = red16(ps2[r]);
            mean[r] = s * (1.f/128.f);
            rstd[r] = rsq_f(s2 * (1.f/128.f) - mean[r]*mean[r] + 1e-5f);
        }
        #pragma unroll
        for (int nt = 0; nt < 8; ++nt) {
            float g = gg[nt*16 + col], b = be[nt*16 + col];
            alignas(16) _Float16 tmp[8];
            #pragma unroll
            for (int r = 0; r < 8; ++r)
                tmp[r] = (_Float16)silu_f((cc[nt][r] - mean[r]) * rstd[r] * g + b);
            *(float4*)&S.hbuf[w][nt*16 + col][rb] = *(const float4*)tmp;
        }
    };

    auto load_A = [&](v16h* a) {
        int kb = hi ? 16 : 0;
        #pragma unroll
        for (int kk = 0; kk < 4; ++kk) {
            #pragma unroll
            for (int j = 0; j < 8; ++j) {
                a[kk][j]     = S.hbuf[w][32*kk + kb + j][rr];
                a[kk][8 + j] = S.hbuf[w][32*kk + kb + 8 + j][rr];
            }
        }
    };

    // ---------------- time loop ---------------------------------------------------
    float minCn = 3.4e38f, sumCn = 0.f;

    for (int t = 0; t < T_SZ - 1; ++t) {
        float tt  = P.tp[t];
        float dtv = P.tp[t+1] - tt;
        float sdt = sqrt_f(fabsf(dtv));

        // A0: x = [z(8), t, treat=0(16), pad] -- also feeds cn/dif (their K>=8 rows are 0)
        v16h a0;
        #pragma unroll
        for (int i = 0; i < 16; ++i) a0[i] = (_Float16)0.f;
        if (hi == 0) {
            #pragma unroll
            for (int k = 0; k < 8; ++k) a0[k] = (_Float16)S.zbuf[w*16 + rr][k];
            a0[8] = (_Float16)tt;
        }

        // ================= cn/dif fused MLP on WMMA =================
        {
            v8f hc[8];
            #pragma unroll
            for (int nt = 0; nt < 8; ++nt) {
                const v16h bfr = *(const v16h*)&S.Bcd1[nt][lane][0];
                hc[nt] = __builtin_amdgcn_wmma_f32_16x16x32_f16(false, a0, false, bfr,
                                                                (short)0, v8zero(), false, false);
            }
            #pragma unroll
            for (int nt = 0; nt < 8; ++nt) {
                float bias = S.cdb1[nt*16 + col];
                alignas(16) _Float16 tmp[8];
                #pragma unroll
                for (int r = 0; r < 8; ++r)
                    tmp[r] = (_Float16)silu_f(hc[nt][r] + bias);
                *(float4*)&S.hbuf[w][nt*16 + col][rb] = *(const float4*)tmp;
            }
            v16h acd[4];
            load_A(acd);
            v8f o2 = v8zero();
            #pragma unroll
            for (int kk = 0; kk < 4; ++kk) {
                const v16h bfr = *(const v16h*)&S.Bcd2[kk][lane][0];
                o2 = __builtin_amdgcn_wmma_f32_16x16x32_f16(false, acd[kk], false, bfr,
                                                            (short)0, o2, false, false);
            }
            if (col < 9) {
                #pragma unroll
                for (int r = 0; r < 8; ++r)
                    S.dbuf[w*16 + rb + r][col] = o2[r];
            }
        }
        // row-lane extraction of cn / dif
        float cn = softplus_f(S.dbuf[zr][0] + S.cnb2);
        float dif[8];
        #pragma unroll
        for (int o = 0; o < 8; ++o)
            dif[o] = softplus_f(S.dbuf[zr][1 + o] + S.difb2[o]) + 0.01f;
        dif[0] *= 0.5f * sqrt_f(cn + 1e-8f);

        // fit (scalar input -> keep on VALU, 2 lanes/row)
        float accf = 0.f;
        for (int jj = 0; jj < 32; ++jj) {
            int j = hi * 32 + jj;
            accf += silu_f(cn * S.fitw1[j] + S.fitb1[j]) * S.fitw2[j];
        }
        accf += __shfl_xor(accf, 16);
        float fit = tanh_f(accf + S.fitb2);

        // ================= drift MLP on WMMA =================
        v8f c[8];
        #pragma unroll
        for (int nt = 0; nt < 8; ++nt) {
            const v16h bfr = *(const v16h*)&S.Bdr1[nt][lane][0];
            c[nt] = __builtin_amdgcn_wmma_f32_16x16x32_f16(false, a0, false, bfr,
                                                           (short)0, v8zero(), false, false);
        }
        ln_silu_store(c, S.b1, S.g1, S.bt1);

        v16h a[4];
        load_A(a);
        #pragma unroll
        for (int nt = 0; nt < 8; ++nt) {
            v8f acc = v8zero();
            #pragma unroll
            for (int kk = 0; kk < 4; ++kk) {
                const v16h bfr = *(const v16h*)&S.Bdr2[kk*8 + nt][lane][0];
                acc = __builtin_amdgcn_wmma_f32_16x16x32_f16(false, a[kk], false, bfr,
                                                             (short)0, acc, false, false);
            }
            c[nt] = acc;
        }
        ln_silu_store(c, S.b2, S.g2, S.bt2);

        load_A(a);
        v8f d3 = v8zero();
        #pragma unroll
        for (int kk = 0; kk < 4; ++kk) {
            const v16h bfr = *(const v16h*)&S.Bdr3[kk][lane][0];
            d3 = __builtin_amdgcn_wmma_f32_16x16x32_f16(false, a[kk], false, bfr,
                                                        (short)0, d3, false, false);
        }
        if (col < 8) {
            #pragma unroll
            for (int r = 0; r < 8; ++r)
                S.dbuf[w*16 + rb + r][col] = d3[r] + S.drb3[col];
        }

        // ================= Euler-Maruyama update =================
        float zz[8];
        #pragma unroll
        for (int k = 0; k < 8; ++k) zz[k] = S.zbuf[zr][k];

        if (hi == 0) {
            sumCn += cn;
            minCn = fminf(minCn, cn);
            P.o_cn[gRow * T_SZ + t] = cn;
            const float4* np = (const float4*)&P.noise[((long long)t * B_SZ + gRow) * 8];
            float4 n0 = np[0], n1 = np[1];
            float nz[8] = { n0.x, n0.y, n0.z, n0.w, n1.x, n1.y, n1.z, n1.w };
            float znv[8];
            #pragma unroll
            for (int k = 0; k < 8; ++k) {
                float dk = S.dbuf[zr][k];
                float drift = (k == 0) ? (dk + 0.1f * fit) * (1.f + growth) : dk;
                znv[k] = zz[k] + drift * dtv + dif[k] * nz[k] * sdt;
                S.zbuf[zr][k] = znv[k];
            }
            float4 zo0 = make_float4(znv[0], znv[1], znv[2], znv[3]);
            float4 zo1 = make_float4(znv[4], znv[5], znv[6], znv[7]);
            float4* tp4 = (float4*)&P.o_traj[(gRow * T_SZ + t + 1) * 8];
            tp4[0] = zo0; tp4[1] = zo1;
        }
    }

    // ---------------- epilogue ----------------------------------------------------
    {
        float zz[8];
        #pragma unroll
        for (int k = 0; k < 8; ++k) zz[k] = S.zbuf[zr][k];
        float cnF = cn_dec(zz);
        if (hi == 0) {
            sumCn += cnF;
            minCn = fminf(minCn, cnF);
            P.o_cn[gRow * T_SZ + (T_SZ - 1)] = cnF;
            P.o_final[gRow] = cnF;
            P.o_ext[gRow]   = sigmoid_f(1.f - minCn);
            P.o_res[gRow]   = 0.f;
            P.o_var[gRow]   = 0.25f * sumCn / (float)T_SZ;
        }
    }
}

extern "C" void kernel_launch(void* const* d_in, const int* in_sizes, int n_in,
                              void* d_out, int out_size, void* d_ws, size_t ws_size,
                              hipStream_t stream) {
    (void)in_sizes; (void)n_in; (void)out_size; (void)d_ws; (void)ws_size;
    KParams P;
    P.init   = (const float*)d_in[0];
    P.tp     = (const float*)d_in[1];
    P.noise  = (const float*)d_in[2];
    P.enc_w1 = (const float*)d_in[3];  P.enc_b1 = (const float*)d_in[4];
    P.enc_g  = (const float*)d_in[5];  P.enc_bt = (const float*)d_in[6];
    P.enc_w2 = (const float*)d_in[7];  P.enc_b2 = (const float*)d_in[8];
    P.cn_w1  = (const float*)d_in[9];  P.cn_b1  = (const float*)d_in[10];
    P.cn_w2  = (const float*)d_in[11]; P.cn_b2  = (const float*)d_in[12];
    P.dr_w1  = (const float*)d_in[13]; P.dr_b1  = (const float*)d_in[14];
    P.dr_g1  = (const float*)d_in[15]; P.dr_bt1 = (const float*)d_in[16];
    P.dr_w2  = (const float*)d_in[17]; P.dr_b2  = (const float*)d_in[18];
    P.dr_g2  = (const float*)d_in[19]; P.dr_bt2 = (const float*)d_in[20];
    P.dr_w3  = (const float*)d_in[21]; P.dr_b3  = (const float*)d_in[22];
    P.dif_w1 = (const float*)d_in[23]; P.dif_b1 = (const float*)d_in[24];
    P.dif_w2 = (const float*)d_in[25]; P.dif_b2 = (const float*)d_in[26];
    P.fit_w1 = (const float*)d_in[27]; P.fit_b1 = (const float*)d_in[28];
    P.fit_w2 = (const float*)d_in[29]; P.fit_b2 = (const float*)d_in[30];
    P.te_w1  = (const float*)d_in[31]; P.te_b1  = (const float*)d_in[32];
    P.te_w2  = (const float*)d_in[33]; P.te_b2  = (const float*)d_in[34];

    float* out = (float*)d_out;
    size_t oT = (size_t)B_SZ * T_SZ * 8;
    P.o_traj  = out;
    P.o_cn    = out + oT;
    P.o_final = P.o_cn + (size_t)B_SZ * T_SZ;
    P.o_ext   = P.o_final + B_SZ;
    P.o_res   = P.o_ext + B_SZ;
    P.o_var   = P.o_res + B_SZ;

    dim3 grid(B_SZ / ROWS), block(BLOCK);
    hipLaunchKernelGGL(sde_fused_kernel, grid, block, sizeof(Smem), stream, P);
}
